// Learned2DRelativeSelfAttention_17471926960863
// MI455X (gfx1250) — compile-verified, matrix operand order checked by reading, not run on hardware
//
#include <hip/hip_runtime.h>
#include <math.h>

// ---------------------------------------------------------------------------
// Types for CDNA5 WMMA (wave32): V_WMMA_F32_16X16X32_BF16
// ---------------------------------------------------------------------------
typedef __bf16 bf16;
typedef __attribute__((ext_vector_type(16))) __bf16 v16bf;
typedef __attribute__((ext_vector_type(8)))  __bf16 v8bf;
typedef __attribute__((ext_vector_type(8)))  float  v8f;
typedef __attribute__((ext_vector_type(4)))  float  v4f;
typedef __attribute__((ext_vector_type(4)))  int    v4i;

#define WMMA_BF16(A, B, C) \
  __builtin_amdgcn_wmma_f32_16x16x32_bf16(false, (A), false, (B), (short)0, (C), false, false)

#define LOG2E 1.44269504088896340736f
#define AS1 __attribute__((address_space(1)))
#define AS3 __attribute__((address_space(3)))

// Problem constants
#define S     1024   // W*H = 32*32 spatial positions
#define NH    8
#define C     64
#define AH    512    // NH*C

// Async global->LDS 16-byte copy (ASYNCcnt-tracked, no VGPR round trip).
static __device__ inline void async_cp16(void* lds, const void* g) {
  __builtin_amdgcn_global_load_async_to_lds_b128(
      (AS1 v4i*)g, (AS3 v4i*)lds, 0, 0);
}

// ---------------------------------------------------------------------------
// Fragment loaders. A/B 16-bit fragment lane layout (ISA 7.12.2):
//   lane L holds row (A) / column-of-B m = L%16;
//   v16bf elements 0..7  = K kb..kb+7,  elements 8..15 = K kb+16..kb+23,
//   where kb = 8*(L/16).  Row-major source with row stride (elements).
// ---------------------------------------------------------------------------
static __device__ inline v16bf frag_bf16(const bf16* __restrict__ base, int stride, int lane) {
  const int m  = lane & 15;
  const int kb = (lane >> 4) << 3;
  const bf16* p = base + m * stride + kb;
  v8bf lo = *(const v8bf*)(p);
  v8bf hi = *(const v8bf*)(p + 16);
  v16bf r;
#pragma unroll
  for (int e = 0; e < 8; ++e) { r[e] = lo[e]; r[e + 8] = hi[e]; }
  return r;
}

static __device__ inline v16bf frag_f32(const float* __restrict__ base, int stride, int lane) {
  const int m  = lane & 15;
  const int kb = (lane >> 4) << 3;
  const float* p = base + m * stride + kb;
  v4f a = *(const v4f*)(p);
  v4f b = *(const v4f*)(p + 4);
  v4f c = *(const v4f*)(p + 16);
  v4f d = *(const v4f*)(p + 20);
  v16bf r;
#pragma unroll
  for (int e = 0; e < 4; ++e) {
    r[e]      = (bf16)a[e];
    r[e + 4]  = (bf16)b[e];
    r[e + 8]  = (bf16)c[e];
    r[e + 12] = (bf16)d[e];
  }
  return r;
}

// Reductions across the 16-lane half that shares one C/D-matrix row set.
static __device__ inline float halfwave_max(float x) {
#pragma unroll
  for (int m = 1; m <= 8; m <<= 1) x = fmaxf(x, __shfl_xor(x, m, 32));
  return x;
}
static __device__ inline float halfwave_sum(float x) {
#pragma unroll
  for (int m = 1; m <= 8; m <<= 1) x += __shfl_xor(x, m, 32);
  return x;
}

// ---------------------------------------------------------------------------
// Kernel 0: positional bias tables.
//   rowb[n][j][l] = sum_p row_emb[l-j+31][p] * w_row[n][p]
//   colb[n][i][k] = sum_p col_emb[k-i+31][p] * w_col[n][p]
// ---------------------------------------------------------------------------
__global__ __launch_bounds__(256) void pos_bias_kernel(
    const float* __restrict__ row_emb, const float* __restrict__ col_emb,
    const float* __restrict__ w_row,   const float* __restrict__ w_col,
    float* __restrict__ rowb, float* __restrict__ colb) {
  int gid = blockIdx.x * 256 + threadIdx.x;   // 0..16383
  int sel = gid >> 13;
  int rem = gid & 8191;                       // n*1024 + j*32 + l
  int n = rem >> 10, j = (rem >> 5) & 31, l = rem & 31;
  const float* emb = sel ? col_emb : row_emb;
  const float* w   = sel ? w_col   : w_row;
  float* dst       = sel ? colb    : rowb;
  int idx = l - j + 31;
  float acc = 0.f;
#pragma unroll 8
  for (int p = 0; p < 64; ++p) acc += emb[idx * 64 + p] * w[n * 64 + p];
  dst[rem] = acc;
}

// ---------------------------------------------------------------------------
// Kernel 0b: V pre-pass. hidden (f32, [b][s][d]) -> bf16 transposed [b][d][s]
// so attention can stage V^T tiles with straight async b128 copies.
// ---------------------------------------------------------------------------
__global__ __launch_bounds__(256) void vcvt_kernel(const float* __restrict__ hidden,
                                                   bf16* __restrict__ v_bf) {
  int gid = blockIdx.x * 256 + threadIdx.x;   // 0..131071, 4 elems each
  int idx = gid << 2;
  v4f h = *(const v4f*)(hidden + idx);
  int d = idx & 63, s = (idx >> 6) & (S - 1), b = idx >> 16;
#pragma unroll
  for (int e = 0; e < 4; ++e)
    v_bf[((b * C + d + e) << 10) + s] = (bf16)h[e];
}

// ---------------------------------------------------------------------------
// Kernel 1: Q/K projections via WMMA.  GEMM M=8192 N=512 K=64, done twice.
// Stores bf16 with layout [b][head][s][d]; Q pre-scaled by 1/sqrt(C).
// ---------------------------------------------------------------------------
__global__ __launch_bounds__(256) void qk_proj_kernel(
    const float* __restrict__ hidden,
    const float* __restrict__ w_q, const float* __restrict__ b_q,
    const float* __restrict__ w_k, const float* __restrict__ b_k,
    bf16* __restrict__ q_bf, bf16* __restrict__ k_bf) {
  int lane = threadIdx.x & 31;
  int wv   = threadIdx.x >> 5;
  int t    = blockIdx.x * 8 + wv;   // 0..32767 tiles (2 * 512 * 32)
  int which = t >> 14;              // 0 = Q, 1 = K
  int rem   = t & 16383;
  int rt = rem >> 5, ct = rem & 31;
  int row0 = rt << 4;               // row over b*S
  int ah0  = ct << 4;               // output column base
  const float* w  = which ? w_k : w_q;
  const float* bb = which ? b_k : b_q;
  bf16* dst   = which ? k_bf : q_bf;
  float scale = which ? 1.0f : 0.125f;   // 1/sqrt(64) folded into Q

  v16bf a0 = frag_f32(hidden + row0 * C,      C, lane);
  v16bf a1 = frag_f32(hidden + row0 * C + 32, C, lane);
  v16bf b0 = frag_f32(w + ah0 * C,      C, lane);   // B col n = w row ah0+n
  v16bf b1 = frag_f32(w + ah0 * C + 32, C, lane);
  v8f acc = {};
  acc = WMMA_BF16(a0, b0, acc);
  acc = WMMA_BF16(a1, b1, acc);

  int n    = lane & 15;
  int mb   = (lane >> 4) << 3;
  float bias = bb[ah0 + n];
  int head = (ah0 + n) >> 6;
  int d    = (ah0 + n) & 63;
#pragma unroll
  for (int r = 0; r < 8; ++r) {
    int row = row0 + mb + r;
    int b   = row >> 10;
    int s   = row & (S - 1);
    dst[(((b * NH + head) * S) + s) * C + d] = (bf16)((acc[r] + bias) * scale);
  }
}

// ---------------------------------------------------------------------------
// Kernel 2: fused flash attention with 2D relative bias.
// Grid: 512 blocks = (b, head, 128-query block). 8 waves x 16-query tiles.
// K and V^T tiles double-buffered in LDS via async global->LDS b128 copies
// (ASYNCcnt): block k+1's tiles are in flight while block k computes.
// ---------------------------------------------------------------------------
__global__ __launch_bounds__(256) void attn_kernel(
    const bf16* __restrict__ q_bf, const bf16* __restrict__ k_bf,
    const bf16* __restrict__ v_bf,
    const float* __restrict__ rowb, const float* __restrict__ colb,
    bf16* __restrict__ o_bf) {
  __shared__ __align__(16) bf16  Kb[2][32 * 64];   // 8 KB  [key][d]
  __shared__ __align__(16) bf16  Vb[2][64 * 32];   // 8 KB  [d][key]
  __shared__ __align__(16) float Rw[32 * 32];      // 4 KB  rowb[head]
  __shared__ __align__(16) float Pl[8 * 16 * 32];  // 16 KB per-wave prob tiles

  int tid  = threadIdx.x;
  int lane = tid & 31;
  int wv   = tid >> 5;
  int bid  = blockIdx.x;
  int b    = bid >> 6;
  int head = (bid >> 3) & 7;
  int qblk = bid & 7;
  int s0   = (qblk << 7) + (wv << 4);   // first query row of this wave
  int bh   = b * NH + head;

  // Stage rowb[head] (32x32 f32) once.
  ((v4f*)Rw)[tid] = ((const v4f*)(rowb + head * 1024))[tid];

  // Per-thread async staging of one key block (32 keys): one b128 of K,
  // one b128 of V^T. Both tiles are contiguous/strided bf16, no conversion.
  const bf16* ksrc = k_bf + bh * S * C;
  const bf16* vsrc = v_bf + b * C * S;
  int vrow = tid >> 2, vseg = (tid & 3) << 3;
#define STAGE(T0, BUF)                                                        \
  do {                                                                        \
    async_cp16(&Kb[BUF][tid << 3], ksrc + (T0) * C + (tid << 3));             \
    async_cp16(&Vb[BUF][vrow * 32 + vseg], vsrc + (vrow << 10) + (T0) + vseg);\
  } while (0)

  // Q fragments held in registers for the whole kernel.
  v16bf qa0 = frag_bf16(q_bf + (bh * S + s0) * C,      C, lane);
  v16bf qa1 = frag_bf16(q_bf + (bh * S + s0) * C + 32, C, lane);

  v8f o0 = {}, o1 = {}, o2 = {}, o3 = {};
  float rm[8], rs[8];
#pragma unroll
  for (int r = 0; r < 8; ++r) { rm[r] = -INFINITY; rs[r] = 0.f; }

  int j0 = s0 & 31;          // query j (second spatial axis) base
  int i0 = s0 >> 5;          // query i (first spatial axis), constant per tile
  int mb = (lane >> 4) << 3; // row base for this half-wave
  int n  = lane & 15;        // column within 16x16 tile
  float* plds = Pl + wv * 512;

  STAGE(0, 0);               // prologue: block 0 in flight

  for (int blk = 0; blk < 32; ++blk) {
    int cur = blk & 1;
    __syncthreads();         // everyone done computing blk-1 -> buf cur^1 free
    if (blk + 1 < 32) {
      STAGE((blk + 1) << 5, cur ^ 1);                    // prefetch next block
      asm volatile("s_wait_asynccnt 0x2" ::: "memory");  // current block landed
    } else {
      asm volatile("s_wait_asynccnt 0x0" ::: "memory");
    }
    __syncthreads();         // all threads' copies of current block visible

    const bf16* Klds = Kb[cur];
    const bf16* Vt   = Vb[cur];
    float cterm = colb[head * 1024 + i0 * 32 + blk];   // colb[n][i0][k0]

    // Scores for two 16-key tiles + bias.
    float sc[2][8];
#pragma unroll
    for (int half = 0; half < 2; ++half) {
      v16bf kb0 = frag_bf16(Klds + (half << 4) * 64,      64, lane);
      v16bf kb1 = frag_bf16(Klds + (half << 4) * 64 + 32, 64, lane);
      v8f s = {};
      s = WMMA_BF16(qa0, kb0, s);
      s = WMMA_BF16(qa1, kb1, s);
      int l0 = half << 4;
#pragma unroll
      for (int r = 0; r < 8; ++r)
        sc[half][r] = s[r] + Rw[(j0 + mb + r) * 32 + l0 + n] + cterm;
    }

    // Online softmax update (rows live across a 16-lane half).
    float fac[8], lsum[8];
#pragma unroll
    for (int r = 0; r < 8; ++r) {
      float lmax = halfwave_max(fmaxf(sc[0][r], sc[1][r]));
      float nm   = fmaxf(rm[r], lmax);
      fac[r] = __builtin_amdgcn_exp2f((rm[r] - nm) * LOG2E);
      rm[r]  = nm;
      float p0 = __builtin_amdgcn_exp2f((sc[0][r] - nm) * LOG2E);
      float p1 = __builtin_amdgcn_exp2f((sc[1][r] - nm) * LOG2E);
      plds[(mb + r) * 32 + n]      = p0;
      plds[(mb + r) * 32 + 16 + n] = p1;
      lsum[r] = p0 + p1;
    }
#pragma unroll
    for (int r = 0; r < 8; ++r) {
      rs[r] = rs[r] * fac[r] + halfwave_sum(lsum[r]);
      o0[r] *= fac[r]; o1[r] *= fac[r]; o2[r] *= fac[r]; o3[r] *= fac[r];
    }

    // Intra-wave cross-lane LDS RAW: wait for the prob stores to land.
    asm volatile("s_wait_dscnt 0x0" ::: "memory");

    // P (16x32) @ V (32x64): 4 output chunks of 16 columns.
    v16bf pa = frag_f32(plds, 32, lane);
    o0 = WMMA_BF16(pa, frag_bf16(Vt +  0 * 32, 32, lane), o0);
    o1 = WMMA_BF16(pa, frag_bf16(Vt + 16 * 32, 32, lane), o1);
    o2 = WMMA_BF16(pa, frag_bf16(Vt + 32 * 32, 32, lane), o2);
    o3 = WMMA_BF16(pa, frag_bf16(Vt + 48 * 32, 32, lane), o3);
  }
#undef STAGE

  // Normalize and store head outputs, bf16 layout [b][s][head*64 + d].
#pragma unroll
  for (int r = 0; r < 8; ++r) {
    float inv = 1.0f / rs[r];
    int s = s0 + mb + r;
    bf16* dst = o_bf + (b * S + s) * AH + head * C;
    dst[n]      = (bf16)(o0[r] * inv);
    dst[16 + n] = (bf16)(o1[r] * inv);
    dst[32 + n] = (bf16)(o2[r] * inv);
    dst[48 + n] = (bf16)(o3[r] * inv);
  }
}

// ---------------------------------------------------------------------------
// Kernel 3: output projection via WMMA. GEMM M=8192 N=64 K=512.
// ---------------------------------------------------------------------------
__global__ __launch_bounds__(256) void out_proj_kernel(
    const bf16* __restrict__ o_bf,
    const float* __restrict__ w_v, const float* __restrict__ b_v,
    float* __restrict__ out) {
  int lane = threadIdx.x & 31;
  int wv   = threadIdx.x >> 5;
  int t    = blockIdx.x * 8 + wv;   // 0..2047 tiles (512 x 4)
  int rt = t >> 2, ct = t & 3;
  int row0 = rt << 4, n0 = ct << 4;

  v8f acc = {};
#pragma unroll
  for (int kc = 0; kc < 16; ++kc) {
    v16bf a  = frag_bf16(o_bf + row0 * AH + kc * 32, AH, lane);
    v16bf bf = frag_f32 (w_v  + n0  * AH + kc * 32, AH, lane);  // B col n = w_v row n0+n
    acc = WMMA_BF16(a, bf, acc);
  }

  int n  = lane & 15;
  int mb = (lane >> 4) << 3;
  float bias = b_v[n0 + n];
#pragma unroll
  for (int r = 0; r < 8; ++r)
    out[(row0 + mb + r) * C + n0 + n] = acc[r] + bias;
}

// ---------------------------------------------------------------------------
// Launch
// ---------------------------------------------------------------------------
extern "C" void kernel_launch(void* const* d_in, const int* in_sizes, int n_in,
                              void* d_out, int out_size, void* d_ws, size_t ws_size,
                              hipStream_t stream) {
  (void)in_sizes; (void)n_in; (void)out_size; (void)ws_size;
  const float* hidden  = (const float*)d_in[0];
  // d_in[1] = attention_mask (unused by the reference forward)
  const float* row_emb = (const float*)d_in[2];
  const float* col_emb = (const float*)d_in[3];
  const float* w_row   = (const float*)d_in[4];
  const float* w_col   = (const float*)d_in[5];
  const float* w_q     = (const float*)d_in[6];
  const float* b_q     = (const float*)d_in[7];
  const float* w_k     = (const float*)d_in[8];
  const float* b_k     = (const float*)d_in[9];
  const float* w_v     = (const float*)d_in[10];
  const float* b_v     = (const float*)d_in[11];
  float* out = (float*)d_out;

  char* ws = (char*)d_ws;
  float* rowb = (float*)(ws);                                   // 32 KB
  float* colb = (float*)(ws + (32u << 10));                     // 32 KB
  bf16*  q_bf = (bf16*)(ws + (64u << 10));                      // 8 MB
  bf16*  k_bf = (bf16*)(ws + (64u << 10) + (8u << 20));         // 8 MB
  bf16*  o_bf = (bf16*)(ws + (64u << 10) + (16u << 20));        // 8 MB
  bf16*  v_bf = (bf16*)(ws + (64u << 10) + (24u << 20));        // 1 MB

  pos_bias_kernel<<<64,   256, 0, stream>>>(row_emb, col_emb, w_row, w_col, rowb, colb);
  vcvt_kernel   <<<512,  256, 0, stream>>>(hidden, v_bf);
  qk_proj_kernel<<<4096, 256, 0, stream>>>(hidden, w_q, b_q, w_k, b_k, q_bf, k_bf);
  attn_kernel   <<<512,  256, 0, stream>>>(q_bf, k_bf, v_bf, rowb, colb, o_bf);
  out_proj_kernel<<<256, 256, 0, stream>>>(o_bf, w_v, b_v, out);
}
